// OwnGCN_73443940761888
// MI455X (gfx1250) — matrix operand order repaired and checked by reading, not compiled, not run on hardware
//
#include <hip/hip_runtime.h>
#include <hip/hip_bf16.h>
#include <stdint.h>

#define N_NODES  50000
#define N_EDGES  800000
#define N_GRAPHS 50

typedef __attribute__((ext_vector_type(16))) __bf16 v16bf;
typedef __attribute__((ext_vector_type(8)))  __bf16 v8bf;
typedef __attribute__((ext_vector_type(8)))  float  v8f;

__device__ __forceinline__ __bf16 f2bf(float f) {
  unsigned u = __builtin_bit_cast(unsigned, f);
  unsigned r = u + 0x7fffu + ((u >> 16) & 1u);   // round-to-nearest-even
  unsigned short h = (unsigned short)(r >> 16);
  return __builtin_bit_cast(__bf16, h);
}

// ---------------------------------------------------------------------------
// WMMA GEMM: C (+)= alpha * A(f32,N x K) @ W(f32,K x Cout)  [+ bias if initC]
// A and W are converted to bf16 in LDS; accumulation in f32 via
// V_WMMA_F32_16X16X32_BF16. Block tile 128x64, 8 wave32s, wave tile 16x64.
// ---------------------------------------------------------------------------
#define BM 128
#define BN 64
#define BK 32

__global__ __launch_bounds__(256)
void wmma_gemm_acc(const float* __restrict__ A, const float* __restrict__ W,
                   float* __restrict__ C, const float* __restrict__ bias,
                   int nrows, int K, int cout, float alpha, int initC)
{
  __shared__ __align__(64) __bf16 Asl[BM * BK];   // row-major [m][k]
  __shared__ __align__(64) __bf16 Bsl[BN * BK];   // col-major [n][k]

  const int tid  = threadIdx.x;
  const int lane = tid & 31;
  const int wave = tid >> 5;                       // 0..7 -> M slab
  const int rowBase = blockIdx.y * BM;
  const int colBase = blockIdx.x * BN;

  v8f acc[4] = {};

  const int ar = tid >> 1;              // 0..127 : A row in tile
  const int ak = (tid & 1) * 16;        // half-row of 32 K
  const int grow = rowBase + ar;
  const int bc = tid >> 2;              // 0..63 : W column in tile
  const int bk = (tid & 3) * 8;         // 8-K chunk

  for (int k0 = 0; k0 < K; k0 += BK) {
    // ---- stage A tile (f32 -> bf16) ----
    if (grow < nrows) {
      const float* s = A + (size_t)grow * K + k0 + ak;
      if (k0 + BK < K) __builtin_prefetch(s + BK, 0, 1);   // global_prefetch_b8
      #pragma unroll
      for (int i = 0; i < 16; i += 4) {
        float4 v = *(const float4*)(s + i);
        __bf16* d = &Asl[ar * BK + ak + i];
        d[0] = f2bf(v.x); d[1] = f2bf(v.y); d[2] = f2bf(v.z); d[3] = f2bf(v.w);
      }
    } else {
      #pragma unroll
      for (int i = 0; i < 16; ++i) Asl[ar * BK + ak + i] = f2bf(0.f);
    }
    // ---- stage B tile transposed: Bsl[n*BK + k] ----
    {
      const float* s = W + (size_t)(k0 + bk) * cout + colBase + bc;
      #pragma unroll
      for (int i = 0; i < 8; ++i)
        Bsl[bc * BK + bk + i] = f2bf(s[(size_t)i * cout]);
    }
    __syncthreads();

    // ---- fragments per ISA 16-bit layouts (wave32) ----
    const int m = lane & 15;
    const int g = lane >> 4;
    const __bf16* arow = &Asl[(wave * 16 + m) * BK];
    v8bf lo = *(const v8bf*)(arow + g * 8);        // K = g*8 .. g*8+7
    v8bf hi = *(const v8bf*)(arow + 16 + g * 8);   // K = 16+g*8 ..
    v16bf a;
    #pragma unroll
    for (int i = 0; i < 8; ++i) { a[i] = lo[i]; a[i + 8] = hi[i]; }

    #pragma unroll
    for (int j = 0; j < 4; ++j) {
      const __bf16* bcol = &Bsl[(j * 16 + m) * BK];
      v8bf blo = *(const v8bf*)(bcol + g * 8);
      v8bf bhi = *(const v8bf*)(bcol + 16 + g * 8);
      v16bf b;
      #pragma unroll
      for (int i = 0; i < 8; ++i) { b[i] = blo[i]; b[i + 8] = bhi[i]; }
      acc[j] = __builtin_amdgcn_wmma_f32_16x16x32_bf16(
                   false, a, false, b, (short)0, acc[j], false, false);
    }
    __syncthreads();
  }

  // ---- store C (f32 16x16 layout: VGPR i -> M = i + 8*g, N = lane&15) ----
  const int n = lane & 15;
  const int g = lane >> 4;
  #pragma unroll
  for (int j = 0; j < 4; ++j) {
    const int col = colBase + j * 16 + n;
    #pragma unroll
    for (int i = 0; i < 8; ++i) {
      const int row = rowBase + wave * 16 + i + 8 * g;
      if (row < nrows) {
        size_t off = (size_t)row * cout + col;
        float v = alpha * acc[j][i];
        if (initC) C[off] = v + (bias ? bias[col] : 0.f);
        else       C[off] += v;
      }
    }
  }
}

// ---------------------------------------------------------------------------
// utility / graph kernels
// ---------------------------------------------------------------------------
__global__ void fill_f32(float* __restrict__ p, float v, long long n) {
  long long i = (long long)blockIdx.x * blockDim.x + threadIdx.x;
  if (i < n) p[i] = v;
}

__global__ void deg_kernel(const int* __restrict__ dst, float* __restrict__ deg, int nE) {
  int e = blockIdx.x * blockDim.x + threadIdx.x;
  if (e < nE) atomicAdd(&deg[dst[e]], 1.0f);
}

__global__ void cnt_kernel(const int* __restrict__ batch, float* __restrict__ cnt, int n) {
  int i = blockIdx.x * blockDim.x + threadIdx.x;
  if (i < n) atomicAdd(&cnt[batch[i]], 1.0f);
}

__global__ void dinv_kernel(float* __restrict__ deg, int n) {
  int i = blockIdx.x * blockDim.x + threadIdx.x;
  if (i < n) {
    float d = deg[i];
    deg[i] = d > 0.f ? rsqrtf(fmaxf(d, 1.f)) : 0.f;   // in-place -> dinv
  }
}

__global__ void nw_kernel(const int* __restrict__ src, const int* __restrict__ dst,
                          const float* __restrict__ dinv, float* __restrict__ nw, int nE) {
  int e = blockIdx.x * blockDim.x + threadIdx.x;
  if (e < nE) nw[e] = -dinv[src[e]] * dinv[dst[e]];
}

__global__ void wsub_kernel(const float* __restrict__ W, float* __restrict__ wmod,
                            int n, int off2) {  // wmod = W0 - W2
  int i = blockIdx.x * blockDim.x + threadIdx.x;
  if (i < n) wmod[i] = W[i] - W[i + off2];
}

// hout[dst] += nw * hin[src], 4 features per thread, atomics land in L2
__global__ void scatter_prop(const float* __restrict__ hin, float* __restrict__ hout,
                             const int* __restrict__ src, const int* __restrict__ dst,
                             const float* __restrict__ nw, int nE, int cfeat, int perShift)
{
  long long idx = (long long)blockIdx.x * blockDim.x + threadIdx.x;
  long long total = (long long)nE << perShift;
  if (idx >= total) return;
  int e = (int)(idx >> perShift);
  int f = (((int)idx) & ((1 << perShift) - 1)) << 2;
  int s = src[e], d = dst[e];
  float w = nw[e];
  float4 v = *(const float4*)(hin + (size_t)s * cfeat + f);
  float* o = hout + (size_t)d * cfeat + f;
  atomicAdd(o + 0, w * v.x);
  atomicAdd(o + 1, w * v.y);
  atomicAdd(o + 2, w * v.z);
  atomicAdd(o + 3, w * v.w);
}

__global__ void gn_sum(const float* __restrict__ h, const int* __restrict__ batch,
                       float* __restrict__ msum, int cfeat) {
  int i = blockIdx.x;
  int b = batch[i];
  const float* hp = h + (size_t)i * cfeat;
  float* mp = msum + (size_t)b * cfeat;
  for (int f = threadIdx.x; f < cfeat; f += blockDim.x)
    atomicAdd(&mp[f], hp[f]);
}

__global__ void gn_pass2(float* __restrict__ h, const int* __restrict__ batch,
                         const float* __restrict__ msum, const float* __restrict__ cnt,
                         const float* __restrict__ ms, float* __restrict__ vsum, int cfeat) {
  int i = blockIdx.x;
  int b = batch[i];
  float inv = 1.f / fmaxf(cnt[b], 1.f);
  float* hp = h + (size_t)i * cfeat;
  const float* mp = msum + (size_t)b * cfeat;
  float* vp = vsum + (size_t)b * cfeat;
  for (int f = threadIdx.x; f < cfeat; f += blockDim.x) {
    float sub = hp[f] - ms[f] * (mp[f] * inv);
    hp[f] = sub;                              // store sub in place
    atomicAdd(&vp[f], sub * sub);
  }
}

__global__ void gn_pass3(float* __restrict__ h, const int* __restrict__ batch,
                         const float* __restrict__ vsum, const float* __restrict__ cnt,
                         const float* __restrict__ gw, const float* __restrict__ gb, int cfeat) {
  int i = blockIdx.x;
  int b = batch[i];
  float inv = 1.f / fmaxf(cnt[b], 1.f);
  float* hp = h + (size_t)i * cfeat;
  const float* vp = vsum + (size_t)b * cfeat;
  for (int f = threadIdx.x; f < cfeat; f += blockDim.x) {
    float var = vp[f] * inv;
    float v = gw[f] * hp[f] * rsqrtf(var + 1e-5f) + gb[f];
    hp[f] = v >= 0.f ? v : 0.2f * v;          // leaky_relu(0.2)
  }
}

__global__ void resid_kernel(float* __restrict__ h, const float* __restrict__ x, long long n) {
  long long i = (long long)blockIdx.x * blockDim.x + threadIdx.x;
  if (i < n) h[i] += x[i];
}

__global__ void pool_sum(const float* __restrict__ h, const int* __restrict__ batch,
                         float* __restrict__ pooled) {
  int i = blockIdx.x;
  int b = batch[i];
  int f = threadIdx.x;  // 128 threads
  atomicAdd(&pooled[(size_t)b * 128 + f], h[(size_t)i * 128 + f]);
}

__global__ void head_kernel(const float* __restrict__ pooled, const float* __restrict__ cnt,
                            const float* __restrict__ W1, const float* __restrict__ b1,
                            const float* __restrict__ W2, const float* __restrict__ b2,
                            float* __restrict__ out) {
  int b = blockIdx.x;
  int j = threadIdx.x;   // 64 threads
  __shared__ float t1[64];
  float inv = 1.f / fmaxf(cnt[b], 1.f);
  float s = b1[j];
  for (int k = 0; k < 128; ++k)
    s += (pooled[(size_t)b * 128 + k] * inv) * W1[k * 64 + j];
  t1[j] = tanhf(s);
  __syncthreads();
  if (j < 10) {
    float o = b2[j];
    for (int k = 0; k < 64; ++k) o += t1[k] * W2[k * 10 + j];
    out[b * 10 + j] = o;
  }
}

// ---------------------------------------------------------------------------
extern "C" void kernel_launch(void* const* d_in, const int* in_sizes, int n_in,
                              void* d_out, int out_size, void* d_ws, size_t ws_size,
                              hipStream_t stream)
{
  const int N = N_NODES, E = N_EDGES, B = N_GRAPHS;
  const float* x    = (const float*)d_in[0];
  const int*   ei   = (const int*)d_in[1];
  const int*   src  = ei;
  const int*   dstp = ei + E;
  const int*   batch = (const int*)d_in[2];
  const float* convW[4] = {(const float*)d_in[3],  (const float*)d_in[8],
                           (const float*)d_in[13], (const float*)d_in[18]};
  const float* convB[4] = {(const float*)d_in[4],  (const float*)d_in[9],
                           (const float*)d_in[14], (const float*)d_in[19]};
  const float* gnw[4]  = {(const float*)d_in[5],  (const float*)d_in[10],
                          (const float*)d_in[15], (const float*)d_in[20]};
  const float* gnb[4]  = {(const float*)d_in[6],  (const float*)d_in[11],
                          (const float*)d_in[16], (const float*)d_in[21]};
  const float* gnms[4] = {(const float*)d_in[7],  (const float*)d_in[12],
                          (const float*)d_in[17], (const float*)d_in[22]};
  const float* lin1W = (const float*)d_in[23];
  const float* lin1b = (const float*)d_in[24];
  const float* lin2W = (const float*)d_in[25];
  const float* lin2b = (const float*)d_in[26];
  float* out = (float*)d_out;

  // workspace carve-up (floats)
  float* ws = (float*)d_ws;
  size_t big = (size_t)N * 512;
  float* buf0   = ws;                         // Tx1 scratch
  float* buf1   = ws + big;
  float* buf2   = ws + 2 * big;
  float* nw     = ws + 3 * big;               // E
  float* deg    = nw + E;                     // N (becomes dinv)
  float* cntB   = deg + N;                    // B
  float* msum   = cntB + B;                   // B*512
  float* vsum   = msum + (size_t)B * 512;     // B*512
  float* pooled = vsum + (size_t)B * 512;     // B*128
  float* wmod   = pooled + (size_t)B * 128;   // up to 512*256

  auto cdiv = [](long long a, long long b) { return (unsigned)((a + b - 1) / b); };

  // ---- graph structure: deg -> dinv, edge weights nw, per-graph counts ----
  fill_f32<<<cdiv(N, 256), 256, 0, stream>>>(deg, 0.f, N);
  fill_f32<<<cdiv(B, 256), 256, 0, stream>>>(cntB, 0.f, B);
  deg_kernel<<<cdiv(E, 256), 256, 0, stream>>>(dstp, deg, E);
  cnt_kernel<<<cdiv(N, 256), 256, 0, stream>>>(batch, cntB, N);
  dinv_kernel<<<cdiv(N, 256), 256, 0, stream>>>(deg, N);
  nw_kernel<<<cdiv(E, 256), 256, 0, stream>>>(src, dstp, deg, nw, E);

  // ---- 4 ChebConv(K=3) + GraphNorm + LeakyReLU layers ----
  // out = Tx0@(W0-W2) + Tx1@W1 + 2*prop(Tx1)@W2 + b   (Tx2 folded away)
  int dims[5] = {128, 256, 512, 256, 128};
  const float* hin = x;
  float* houts[4] = {buf2, buf1, buf2, buf1};
  float* p2s[4]   = {buf1, buf2, buf1, buf2};   // P2 reuses hin's buffer (l>0)

  for (int l = 0; l < 4; ++l) {
    int cin = dims[l], cout = dims[l + 1];
    float* hout = houts[l];
    float* p2   = p2s[l];
    float* tx1  = buf0;
    int wn = cin * cout;

    wsub_kernel<<<cdiv(wn, 256), 256, 0, stream>>>(convW[l], wmod, wn, 2 * wn);

    long long ncin = (long long)N * cin;
    int ps = (cin == 128) ? 5 : (cin == 256) ? 6 : 7;   // log2(cin/4)
    long long sc = (long long)E << ps;

    fill_f32<<<cdiv(ncin, 256), 256, 0, stream>>>(tx1, 0.f, ncin);
    scatter_prop<<<cdiv(sc, 256), 256, 0, stream>>>(hin, tx1, src, dstp, nw, E, cin, ps);

    dim3 gg(cout / 64, cdiv(N, 128));
    wmma_gemm_acc<<<gg, 256, 0, stream>>>(hin, wmod, hout, convB[l], N, cin, cout, 1.f, 1);
    wmma_gemm_acc<<<gg, 256, 0, stream>>>(tx1, convW[l] + wn, hout, nullptr, N, cin, cout, 1.f, 0);

    fill_f32<<<cdiv(ncin, 256), 256, 0, stream>>>(p2, 0.f, ncin);
    scatter_prop<<<cdiv(sc, 256), 256, 0, stream>>>(tx1, p2, src, dstp, nw, E, cin, ps);
    wmma_gemm_acc<<<gg, 256, 0, stream>>>(p2, convW[l] + 2 * wn, hout, nullptr, N, cin, cout, 2.f, 0);

    long long bc = (long long)B * cout;
    fill_f32<<<cdiv(bc, 256), 256, 0, stream>>>(msum, 0.f, bc);
    fill_f32<<<cdiv(bc, 256), 256, 0, stream>>>(vsum, 0.f, bc);
    gn_sum  <<<N, 256, 0, stream>>>(hout, batch, msum, cout);
    gn_pass2<<<N, 256, 0, stream>>>(hout, batch, msum, cntB, gnms[l], vsum, cout);
    gn_pass3<<<N, 256, 0, stream>>>(hout, batch, vsum, cntB, gnw[l], gnb[l], cout);

    hin = hout;
  }

  // ---- residual + mean-pool + MLP head ----
  float* hf = houts[3];
  long long nf = (long long)N * 128;
  resid_kernel<<<cdiv(nf, 256), 256, 0, stream>>>(hf, x, nf);
  fill_f32<<<cdiv((long long)B * 128, 256), 256, 0, stream>>>(pooled, 0.f, (long long)B * 128);
  pool_sum<<<N, 128, 0, stream>>>(hf, batch, pooled);
  head_kernel<<<B, 64, 0, stream>>>(pooled, cntB, lin1W, lin1b, lin2W, lin2b, out);
}
